// DynamicsMechanism_52226802319586
// MI455X (gfx1250) — matrix operand order repaired and checked by reading, not compiled, not run on hardware
//
#include <hip/hip_runtime.h>
#include <hip/hip_bf16.h>
#include <math.h>

// Problem constants (B,N,D,H) = (32,64,128,256)
#define GB 32
#define GN 64
#define GD 128
#define GH 256
#define ROWS (GB*GN)           // 2048
#define LN_EPS 1e-5f

typedef __attribute__((ext_vector_type(16))) __bf16 v16bf;
typedef __attribute__((ext_vector_type(8)))  float  v8f;

// ---------------------------------------------------------------------------
// Fragment loaders for V_WMMA_F32_16X16X32_BF16 (wave32).
// A (16x32, 16-bit): lane l<16 -> row M=l, K in {0..7, 16..23}; lane l+16 ->
//   same row, K in {8..15, 24..31}; packed 2 per VGPR. Two 16B chunks.
// B (32x16, 16-bit): stored as B^T rows (one row per output column n);
//   lane n<16 -> K in {0..7, 16..23}, lane n+16 -> K in {8..15, 24..31}; two 16B chunks.
// ---------------------------------------------------------------------------
static __device__ __forceinline__ v16bf load_frag_a(const __bf16* __restrict__ row,
                                                    int k0, int half) {
  union { uint4 u[2]; v16bf v; } f;
  const uint4* p = reinterpret_cast<const uint4*>(row + k0 + half * 8);
  f.u[0] = p[0];   // K = k0+8h .. +7
  f.u[1] = p[2];   // K = k0+8h+16 .. +23   (+2 uint4 = +16 bf16 elements)
  return f.v;
}

static __device__ __forceinline__ v16bf load_frag_b(const __bf16* __restrict__ row,
                                                    int k0, int half) {
  union { uint4 u[2]; v16bf v; } f;
  const uint4* p = reinterpret_cast<const uint4*>(row + k0 + half * 8);
  f.u[0] = p[0];   // K = k0+8h .. +7
  f.u[1] = p[2];   // K = k0+8h+16 .. +23
  return f.v;
}

// ---------------------------------------------------------------------------
// Generic bf16 WMMA GEMM: Out[m, colOff+n] = sum_k A[m,k] * Bt[n,k]  (+ epi)
// One wave per 16x16 output tile, 8 waves / 256-thread block.
// Epilogue: + (rowScale ? rowScale[m] : 1) * bias[n]  + resid[m,n]
// Stores f32 (outF) or bf16 (outB).
// ---------------------------------------------------------------------------
__global__ __launch_bounds__(256)
void gemm_wmma_bf16(const __bf16* __restrict__ A,
                    const __bf16* __restrict__ Bt,
                    int M, int N, int K,
                    float* __restrict__ outF, __bf16* __restrict__ outB,
                    int ldOut, int colOff,
                    const float* __restrict__ bias,
                    const float* __restrict__ rowScale,
                    const float* __restrict__ resid, int ldResid)
{
  const int lane = threadIdx.x & 31;
  const int wave = threadIdx.x >> 5;
  const int tilesN = N >> 4;
  const int tile = blockIdx.x * 8 + wave;
  if (tile >= (M >> 4) * tilesN) return;
  const int tm = tile / tilesN;
  const int tn = tile - tm * tilesN;
  const int l    = lane & 15;
  const int half = lane >> 4;

  const __bf16* aRow = A  + (size_t)(tm * 16 + l) * K;
  const __bf16* bRow = Bt + (size_t)(tn * 16 + l) * K;

  v8f acc = {};
  for (int k0 = 0; k0 < K; k0 += 32) {
    v16bf af = load_frag_a(aRow, k0, half);
    v16bf bf = load_frag_b(bRow, k0, half);
    acc = __builtin_amdgcn_wmma_f32_16x16x32_bf16(
        /*neg_a=*/false, af, /*neg_b=*/false, bf,
        /*c_mod=*/(short)0, acc, /*reuse_a=*/false, /*reuse_b=*/false);
  }

  // C/D layout: lane l (N=tn*16+l for both halves); VGPR r -> M = r + 8*half.
  const int n = tn * 16 + l;
  const float bi = bias ? bias[n] : 0.0f;
#pragma unroll
  for (int r = 0; r < 8; ++r) {
    const int m = tm * 16 + half * 8 + r;
    float v = acc[r];
    float b = bi;
    if (rowScale) b *= rowScale[m];
    v += b;
    if (resid) v += resid[(size_t)m * ldResid + n];
    const size_t o = (size_t)m * ldOut + colOff + n;
    if (outB) outB[o] = (__bf16)v;
    else      outF[o] = v;
  }
}

// ---------------------------------------------------------------------------
// Pack / transpose weights and slots to bf16 layouts.
//  Wp   [512][128] : n<256 -> We1[k][n] (top half), n>=256 -> We1[128+k][n-256]
//  We2t [128][256] : We2[k][n]
//  Wn1t [256][256] : Wn1[k][n]
//  Wn2t [128][256] : Wn2[k][n]
//  node_in[:,0:128] = bf16(slots)
// ---------------------------------------------------------------------------
__global__ void pack_kernel(const float* __restrict__ slots,
                            const float* __restrict__ We1,
                            const float* __restrict__ We2,
                            const float* __restrict__ Wn1,
                            const float* __restrict__ Wn2,
                            __bf16* __restrict__ slots_bf,
                            __bf16* __restrict__ node_in,
                            __bf16* __restrict__ Wp,
                            __bf16* __restrict__ We2t,
                            __bf16* __restrict__ Wn1t,
                            __bf16* __restrict__ Wn2t)
{
  const int idx = blockIdx.x * blockDim.x + threadIdx.x;
  if (idx < ROWS * GD) {
    const float v = slots[idx];
    slots_bf[idx] = (__bf16)v;
    const int r = idx >> 7, c = idx & 127;
    node_in[r * (2 * GD) + c] = (__bf16)v;
  }
  if (idx < 512 * GD) {
    const int n = idx >> 7, k = idx & 127;
    const float v = (n < GH) ? We1[k * GH + n]
                             : We1[(GD + k) * GH + (n - GH)];
    Wp[idx] = (__bf16)v;
  }
  if (idx < GD * GH) {               // 128 x 256
    const int n = idx >> 8, k = idx & 255;
    We2t[idx] = (__bf16)We2[k * GD + n];
    Wn2t[idx] = (__bf16)Wn2[k * GD + n];
  }
  if (idx < GH * GH) {               // 256 x 256
    const int n = idx >> 8, k = idx & 255;
    Wn1t[idx] = (__bf16)Wn1[k * GH + n];
  }
}

static __device__ __forceinline__ float gelu_exact(float a) {
  return 0.5f * a * (1.0f + erff(a * 0.70710678118654752f));
}

// ---------------------------------------------------------------------------
// Edge hot loop. One block per (b,i). 8 waves; wave w handles j = w, w+8, ...
// Lane owns 8 strided channels ch = lane + 32*t of H=256.
//   h = P[b,i] + be1 + Q[b,j]; LayerNorm(ge,bge); GELU; S += w_ij * gelu.
// LayerNorm reductions are intra-wave shfl_xor (wave32) -> no barriers.
// Cross-wave S merge via LDS float atomics. S stored bf16 for next WMMA GEMM.
// ---------------------------------------------------------------------------
__global__ __launch_bounds__(256)
void edge_kernel(const float* __restrict__ PQ,      // [ROWS][512] : P | Q
                 const float* __restrict__ adj,     // [B][N][N]
                 const float* __restrict__ be1,
                 const float* __restrict__ ge,
                 const float* __restrict__ bge,
                 __bf16* __restrict__ S,            // [ROWS][256]
                 float* __restrict__ wsum)          // [ROWS]
{
  const int bi   = blockIdx.x;          // b*N + i
  const int b    = bi >> 6;
  const int i    = bi & 63;
  const int lane = threadIdx.x & 31;
  const int wave = threadIdx.x >> 5;

  float p[8], g8[8], bb[8], s[8];
  const float* prow = PQ + (size_t)bi * 512;
#pragma unroll
  for (int t = 0; t < 8; ++t) {
    const int ch = lane + 32 * t;
    p[t]  = prow[ch] + be1[ch];
    g8[t] = ge[ch];
    bb[t] = bge[ch];
    s[t]  = 0.0f;
  }
  float wsm = 0.0f;

  const float* arow = adj + (size_t)bi * GN;
  for (int j = wave; j < GN; j += 8) {
    float w = arow[j];
    if (j == i) w = 0.0f;
    const float* qrow = PQ + (size_t)(b * GN + j) * 512 + GH;
    float h[8];
    float sum = 0.0f, sq = 0.0f;
#pragma unroll
    for (int t = 0; t < 8; ++t) {
      const float q = qrow[lane + 32 * t];
      const float hv = p[t] + q;
      h[t] = hv;
      sum += hv;
      sq  += hv * hv;
    }
#pragma unroll
    for (int off = 16; off; off >>= 1) {
      sum += __shfl_xor(sum, off);
      sq  += __shfl_xor(sq,  off);
    }
    const float mean = sum * (1.0f / 256.0f);
    const float var  = sq * (1.0f / 256.0f) - mean * mean;
    const float rinv = rsqrtf(var + LN_EPS);
#pragma unroll
    for (int t = 0; t < 8; ++t) {
      const float a = (h[t] - mean) * rinv * g8[t] + bb[t];
      s[t] += w * gelu_exact(a);
    }
    wsm += w;
  }

  __shared__ float sS[GH];
  __shared__ float sw;
  sS[threadIdx.x] = 0.0f;
  if (threadIdx.x == 0) sw = 0.0f;
  __syncthreads();
#pragma unroll
  for (int t = 0; t < 8; ++t) atomicAdd(&sS[lane + 32 * t], s[t]);
  if (lane == 0) atomicAdd(&sw, wsm);
  __syncthreads();
  S[(size_t)bi * GH + threadIdx.x] = (__bf16)sS[threadIdx.x];
  if (threadIdx.x == 0) wsum[bi] = sw;
}

// ---------------------------------------------------------------------------
// Node-path LayerNorm + GELU over rows of 256 (bias already added by GEMM).
// One wave per row; lane owns 8 strided channels. Output bf16 for next GEMM.
// ---------------------------------------------------------------------------
__global__ __launch_bounds__(256)
void ln_gelu_rows(const float* __restrict__ X,     // [ROWS][256]
                  const float* __restrict__ g,
                  const float* __restrict__ bgv,
                  __bf16* __restrict__ Y)          // [ROWS][256]
{
  const int lane = threadIdx.x & 31;
  const int wave = threadIdx.x >> 5;
  const int row  = blockIdx.x * 8 + wave;
  const float* xr = X + (size_t)row * GH;

  float h[8];
  float sum = 0.0f, sq = 0.0f;
#pragma unroll
  for (int t = 0; t < 8; ++t) {
    h[t] = xr[lane + 32 * t];
    sum += h[t];
    sq  += h[t] * h[t];
  }
#pragma unroll
  for (int off = 16; off; off >>= 1) {
    sum += __shfl_xor(sum, off);
    sq  += __shfl_xor(sq,  off);
  }
  const float mean = sum * (1.0f / 256.0f);
  const float var  = sq * (1.0f / 256.0f) - mean * mean;
  const float rinv = rsqrtf(var + LN_EPS);

  __bf16* yr = Y + (size_t)row * GH;
#pragma unroll
  for (int t = 0; t < 8; ++t) {
    const int ch = lane + 32 * t;
    const float a = (h[t] - mean) * rinv * g[ch] + bgv[ch];
    yr[ch] = (__bf16)gelu_exact(a);
  }
}

// ---------------------------------------------------------------------------
extern "C" void kernel_launch(void* const* d_in, const int* in_sizes, int n_in,
                              void* d_out, int out_size, void* d_ws, size_t ws_size,
                              hipStream_t stream) {
  (void)in_sizes; (void)n_in; (void)out_size; (void)ws_size;
  const float* slots = (const float*)d_in[0];
  const float* adj   = (const float*)d_in[1];
  const float* We1   = (const float*)d_in[2];
  const float* be1   = (const float*)d_in[3];
  const float* ge    = (const float*)d_in[4];
  const float* bge   = (const float*)d_in[5];
  const float* We2   = (const float*)d_in[6];
  const float* be2   = (const float*)d_in[7];
  const float* Wn1   = (const float*)d_in[8];
  const float* bn1   = (const float*)d_in[9];
  const float* gn    = (const float*)d_in[10];
  const float* bgn   = (const float*)d_in[11];
  const float* Wn2   = (const float*)d_in[12];
  const float* bn2   = (const float*)d_in[13];
  float* out = (float*)d_out;

  char* ws = (char*)d_ws;
  size_t off = 0;
  auto alloc = [&](size_t bytes) -> void* {
    void* p = ws + off;
    off += (bytes + 255) & ~(size_t)255;
    return p;
  };
  __bf16* slots_bf = (__bf16*)alloc((size_t)ROWS * GD * 2);
  __bf16* Wp       = (__bf16*)alloc((size_t)512 * GD * 2);
  __bf16* We2t     = (__bf16*)alloc((size_t)GD * GH * 2);
  __bf16* Wn1t     = (__bf16*)alloc((size_t)GH * GH * 2);
  __bf16* Wn2t     = (__bf16*)alloc((size_t)GD * GH * 2);
  float*  PQ       = (float*)alloc((size_t)ROWS * 512 * 4);
  __bf16* Sbf      = (__bf16*)alloc((size_t)ROWS * GH * 2);
  float*  wsum     = (float*)alloc((size_t)ROWS * 4);
  __bf16* node_in  = (__bf16*)alloc((size_t)ROWS * 2 * GD * 2);
  float*  hn       = (float*)alloc((size_t)ROWS * GH * 4);
  __bf16* an       = (__bf16*)alloc((size_t)ROWS * GH * 2);

  // 1) bf16 packing + weight transposes
  pack_kernel<<<(ROWS * GD) / 256, 256, 0, stream>>>(
      slots, We1, We2, Wn1, Wn2, slots_bf, node_in, Wp, We2t, Wn1t, Wn2t);

  // 2) PQ = slots @ [We1_top | We1_bot]   (M=2048, N=512, K=128)
  gemm_wmma_bf16<<<(ROWS / 16) * (512 / 16) / 8, 256, 0, stream>>>(
      slots_bf, Wp, ROWS, 512, GD,
      PQ, nullptr, 512, 0, nullptr, nullptr, nullptr, 0);

  // 3) per-pair LN + GELU + adjacency-weighted H-space accumulation
  edge_kernel<<<ROWS, 256, 0, stream>>>(PQ, adj, be1, ge, bge, Sbf, wsum);

  // 4) messages = S @ We2 + wsum*be2  -> bf16 into node_in[:,128:256]
  gemm_wmma_bf16<<<(ROWS / 16) * (GD / 16) / 8, 256, 0, stream>>>(
      Sbf, We2t, ROWS, GD, GH,
      nullptr, node_in, 2 * GD, GD, be2, wsum, nullptr, 0);

  // 5) hn = node_in @ Wn1 + bn1        (M=2048, N=256, K=256)
  gemm_wmma_bf16<<<(ROWS / 16) * (GH / 16) / 8, 256, 0, stream>>>(
      node_in, Wn1t, ROWS, GH, 2 * GD,
      hn, nullptr, GH, 0, bn1, nullptr, nullptr, 0);

  // 6) node LayerNorm + GELU
  ln_gelu_rows<<<ROWS / 8, 256, 0, stream>>>(hn, gn, bgn, an);

  // 7) out = slots + an @ Wn2 + bn2    (M=2048, N=128, K=256)
  gemm_wmma_bf16<<<(ROWS / 16) * (GD / 16) / 8, 256, 0, stream>>>(
      an, Wn2t, ROWS, GD, GH,
      out, nullptr, GD, 0, bn2, nullptr, slots, GD);
}
